// FullAttention_74208444940609
// MI455X (gfx1250) — matrix operand order repaired and checked by reading, not compiled, not run on hardware
//
#include <hip/hip_runtime.h>
#include <math.h>
#include <stdint.h>

typedef __attribute__((ext_vector_type(16))) __bf16 v16bf;
typedef __attribute__((ext_vector_type(8)))  float  v8f;
typedef __attribute__((ext_vector_type(4)))  int    v4i;

#define BSZ   2
#define SEQ   2048
#define DM    1024
#define NH    16
#define HD    64

// ---------------------------------------------------------------------------
// CDNA5 async global->LDS copy (ASYNCcnt-tracked), with synchronous fallback.
// Builtin signature (from hipcc diagnostic): (v4i as1*, v4i as3*, imm, imm).
// Generic->AS casts done via integer: for LDS, generic addr[31:0] is the LDS
// offset (ISA 10.2 aperture mapping); for global, generic == global address.
#if defined(__HIP_DEVICE_COMPILE__) && \
    __has_builtin(__builtin_amdgcn_global_load_async_to_lds_b128) && \
    __has_builtin(__builtin_amdgcn_s_wait_asynccnt)
typedef __attribute__((address_space(1))) v4i* gv4i_p;
typedef __attribute__((address_space(3))) v4i* lv4i_p;
#define ASYNC_CP16(g, l)                                                     \
  __builtin_amdgcn_global_load_async_to_lds_b128(                            \
      (gv4i_p)(uintptr_t)(g), (lv4i_p)(unsigned)(uintptr_t)(l), 0, 0)
#define WAIT_ASYNC(n) __builtin_amdgcn_s_wait_asynccnt(n)
#else
#define ASYNC_CP16(g, l) (*(int4*)(l) = *(const int4*)(g))
#define WAIT_ASYNC(n) ((void)0)
#endif

// ---------------------------------------------------------------------------
// Fragment loaders (CDNA5 WMMA 16x16x32 bf16 layouts, ISA 7.12.2)
// A (16x32, MxK): lanes 0-15 & 16-31 hold row M=lane&15.
//   elems 0..7  -> K = (lane>=16 ? 8:0) + 0..7
//   elems 8..15 -> K = (lane>=16 ? 8:0) + 16..23
__device__ __forceinline__ v16bf frag_a_load(const __bf16* rowptr, int lane) {
  const __bf16* p = rowptr + ((lane >> 4) & 1) * 8;
  v16bf a;
#pragma unroll
  for (int i = 0; i < 8; ++i) { a[i] = p[i]; a[i + 8] = p[i + 16]; }
  return a;
}

// B (32x16, KxN): lane holds column n=lane&15; elems 0..7 -> K = 8*(lane>>4) + 0..7, elems 8..15 -> K = 8*(lane>>4) + 16..23.
__device__ __forceinline__ v16bf frag_b_load(const __bf16* colptr, int lane) {
  const __bf16* p = colptr + ((lane >> 4) & 1) * 8;
  v16bf b;
#pragma unroll
  for (int i = 0; i < 8; ++i) { b[i] = p[i]; b[i + 8] = p[i + 16]; }
  return b;
}

// ---------------------------------------------------------------------------
// Shared WMMA GEMM mainloop: acc[4] (16x64 tile per wave) += A[16xDM] * W^T.
// B tile (64 output features x 32 k) is staged in LDS, double-buffered via
// async global->LDS copies; A fragment is software-pipelined in registers.
__device__ __forceinline__ void wmma_gemm64(
    const __bf16* __restrict__ Arow,   // A + arow*DM (this lane's A row)
    const __bf16* __restrict__ wsrc,   // this thread's staging source row
    __bf16 (&Wlds)[2][64][32],
    v8f (&acc)[4], int lane, int srow, int shalf)
{
  // preload buffer 0 (k = 0..31)
  ASYNC_CP16(wsrc + 0, &Wlds[0][srow][shalf * 16]);
  ASYNC_CP16(wsrc + 8, &Wlds[0][srow][shalf * 16 + 8]);
  v16bf a_cur = frag_a_load(Arow, lane);

  int cur = 0;
  for (int kb = 0; kb < DM; kb += 32) {
    const bool hasnext = (kb + 32) < DM;
    if (hasnext) {                       // stage next k-tile into other buffer
      ASYNC_CP16(wsrc + kb + 32, &Wlds[cur ^ 1][srow][shalf * 16]);
      ASYNC_CP16(wsrc + kb + 40, &Wlds[cur ^ 1][srow][shalf * 16 + 8]);
    }
    v16bf a_nxt = {};
    if (hasnext) a_nxt = frag_a_load(Arow + kb + 32, lane);

    if (hasnext) WAIT_ASYNC(2);          // newest 2 = next buf; current done
    else         WAIT_ASYNC(0);
    __syncthreads();                     // publish current buffer

#pragma unroll
    for (int j = 0; j < 4; ++j) {
      v16bf b = frag_b_load(&Wlds[cur][16 * j + (lane & 15)][0], lane);
      acc[j] = __builtin_amdgcn_wmma_f32_16x16x32_bf16(
          false, a_cur, false, b, (short)0, acc[j], false, false);
    }
    __syncthreads();                     // done reading before next overwrite
    a_cur = a_nxt;
    cur ^= 1;
  }
}

// ---------------------------------------------------------------------------
// K0: fp32 -> bf16 conversion
__global__ void cvt_f32_bf16(const float* __restrict__ src,
                             __bf16* __restrict__ dst, int n) {
  int i = blockIdx.x * blockDim.x + threadIdx.x;
  if (i < n) dst[i] = (__bf16)src[i];
}

// ---------------------------------------------------------------------------
// K1: fused QKV projection + RoPE.  grid = (Mtiles=64, heads=16, proj=3)
// q[m,d] = sum_e x[m,e] * W[d,e]   (einsum bse,de->bsd)
__global__ __launch_bounds__(128) void qkv_proj(
    const __bf16* __restrict__ xbf,
    const __bf16* __restrict__ Wqb, const __bf16* __restrict__ Wkb,
    const __bf16* __restrict__ Wvb,
    const float* __restrict__ rcos, const float* __restrict__ rsin,
    __bf16* __restrict__ Qr, __bf16* __restrict__ Kr, __bf16* __restrict__ Vt)
{
  __shared__ __align__(16) __bf16 Wlds[2][64][32];
  const int tid    = threadIdx.x;
  const int lane   = tid & 31;
  const int wave   = tid >> 5;
  const int p      = blockIdx.z;            // 0=Q 1=K 2=V
  const int h      = blockIdx.y;
  const int m_base = blockIdx.x * 64 + wave * 16;
  const int nb     = h * HD;
  const __bf16* W  = (p == 0) ? Wqb : (p == 1) ? Wkb : Wvb;

  const int srow  = tid >> 1;               // 0..63 staged weight row
  const int shalf = tid & 1;                // which 16-elem half of k-tile
  const __bf16* wsrc = W + (size_t)(nb + srow) * DM + shalf * 16;
  const int arow = m_base + (lane & 15);

  v8f acc[4] = {};
  wmma_gemm64(xbf + (size_t)arow * DM, wsrc, Wlds, acc, lane, srow, shalf);

  const int half = (lane >> 4) & 1;
#pragma unroll
  for (int r = 0; r < 8; ++r) {
    int m  = m_base + r + 8 * half;
    int bb = m >> 11;           // / SEQ
    int s  = m & (SEQ - 1);
#pragma unroll
    for (int j = 0; j < 4; ++j) {
      int   dd = (lane & 15) + 16 * j;
      float v  = acc[j][r];
      if (p == 2) {
        // V stored transposed: [b, h, dd, s]
        Vt[(((size_t)bb * NH + h) * HD + dd) * SEQ + s] = (__bf16)v;
      } else {
        // RoPE: rotate-half pairs dd <-> dd+32 == acc tiles j <-> j+2, same lane
        float c   = rcos[s * HD + dd];
        float sn  = rsin[s * HD + dd];
        float rot = (j < 2) ? -acc[j + 2][r] : acc[j - 2][r];
        float o   = v * c + rot * sn;
        __bf16* dst = (p == 0) ? Qr : Kr;
        dst[(((size_t)bb * NH + h) * SEQ + s) * HD + dd] = (__bf16)o;
      }
    }
  }
}

// ---------------------------------------------------------------------------
// K2: flash attention (causal).  grid = (S/64=32, B*H=32), 4 waves x 16 q-rows
__global__ __launch_bounds__(128) void attn(
    const __bf16* __restrict__ Qr, const __bf16* __restrict__ Kr,
    const __bf16* __restrict__ Vt, __bf16* __restrict__ Obf)
{
  __shared__ __bf16 Plds[4][16][72];       // per-wave P staging (C->A transpose)
  const int lane = threadIdx.x & 31;
  const int wave = threadIdx.x >> 5;
  const int bh   = blockIdx.y;
  const int b    = bh >> 4, h = bh & 15;
  const int qw   = blockIdx.x * 64 + wave * 16;
  const int half = (lane >> 4) & 1;

  const __bf16* Qh = Qr + (size_t)bh * SEQ * HD;
  const __bf16* Kh = Kr + (size_t)bh * SEQ * HD;
  const __bf16* Vh = Vt + (size_t)bh * HD * SEQ;

  v16bf qf[2];
  qf[0] = frag_a_load(Qh + (size_t)(qw + (lane & 15)) * HD + 0,  lane);
  qf[1] = frag_a_load(Qh + (size_t)(qw + (lane & 15)) * HD + 32, lane);

  v8f   oacc[4] = {};
  float mrun[8], lrun[8];
#pragma unroll
  for (int r = 0; r < 8; ++r) { mrun[r] = -1e30f; lrun[r] = 0.f; }

  const int nkb = blockIdx.x + 1;          // causal: key blocks 0..blockIdx.x
  for (int kbi = 0; kbi < nkb; ++kbi) {
    const int key0 = kbi * 64;

    // ---- S = Q K^T (16 x 64 tile), 8 WMMAs
    v8f sacc[4] = {};
#pragma unroll
    for (int j = 0; j < 4; ++j) {
      const __bf16* kc = Kh + (size_t)(key0 + 16 * j + (lane & 15)) * HD;
      v16bf b0 = frag_b_load(kc + 0, lane);
      sacc[j] = __builtin_amdgcn_wmma_f32_16x16x32_bf16(
          false, qf[0], false, b0, (short)0, sacc[j], false, false);
      v16bf b1 = frag_b_load(kc + 32, lane);
      sacc[j] = __builtin_amdgcn_wmma_f32_16x16x32_bf16(
          false, qf[1], false, b1, (short)0, sacc[j], false, false);
    }

    // ---- hoist V fragment loads: latency hides under softmax VALU work
    v16bf vf[4][2];
#pragma unroll
    for (int jd = 0; jd < 4; ++jd) {
      const __bf16* vc = Vh + (size_t)((lane & 15) + 16 * jd) * SEQ + key0;
      vf[jd][0] = frag_b_load(vc + 0,  lane);
      vf[jd][1] = frag_b_load(vc + 32, lane);
    }

    // ---- scale, causal mask, online softmax (16-lane row reductions)
    float corr[8];
#pragma unroll
    for (int r = 0; r < 8; ++r) {
      int   q  = qw + r + 8 * half;
      float rm = -1e30f;
#pragma unroll
      for (int j = 0; j < 4; ++j) {
        int   key = key0 + 16 * j + (lane & 15);
        float sv  = sacc[j][r] * 0.125f;           // D^-0.5
        if (key > q) sv = -1e30f;
        sacc[j][r] = sv;
        rm = fmaxf(rm, sv);
      }
#pragma unroll
      for (int msk = 8; msk >= 1; msk >>= 1)
        rm = fmaxf(rm, __shfl_xor(rm, msk, 32));
      float mn = fmaxf(mrun[r], rm);
      corr[r]  = __expf(mrun[r] - mn);
      mrun[r]  = mn;
      float rs = 0.f;
#pragma unroll
      for (int j = 0; j < 4; ++j) {
        float pv = __expf(sacc[j][r] - mn);
        sacc[j][r] = pv;
        rs += pv;
      }
#pragma unroll
      for (int msk = 8; msk >= 1; msk >>= 1)
        rs += __shfl_xor(rs, msk, 32);
      lrun[r] = lrun[r] * corr[r] + rs;
    }

    // ---- P: C-layout -> bf16 A-layout via LDS
#pragma unroll
    for (int r = 0; r < 8; ++r)
#pragma unroll
      for (int j = 0; j < 4; ++j)
        Plds[wave][r + 8 * half][(lane & 15) + 16 * j] = (__bf16)sacc[j][r];
    __syncthreads();
    v16bf pf[2];
    pf[0] = frag_a_load(&Plds[wave][lane & 15][0],  lane);
    pf[1] = frag_a_load(&Plds[wave][lane & 15][32], lane);

    // ---- rescale running O, then O += P V  (8 WMMAs)
#pragma unroll
    for (int j = 0; j < 4; ++j)
#pragma unroll
      for (int r = 0; r < 8; ++r) oacc[j][r] *= corr[r];
#pragma unroll
    for (int jd = 0; jd < 4; ++jd) {
      oacc[jd] = __builtin_amdgcn_wmma_f32_16x16x32_bf16(
          false, pf[0], false, vf[jd][0], (short)0, oacc[jd], false, false);
      oacc[jd] = __builtin_amdgcn_wmma_f32_16x16x32_bf16(
          false, pf[1], false, vf[jd][1], (short)0, oacc[jd], false, false);
    }
    __syncthreads();
  }

  // ---- normalize, store O as bf16 in [b, s, h*64+dd]
#pragma unroll
  for (int r = 0; r < 8; ++r) {
    int   q   = qw + r + 8 * half;
    float inv = 1.0f / lrun[r];
#pragma unroll
    for (int jd = 0; jd < 4; ++jd) {
      int dd = (lane & 15) + 16 * jd;
      Obf[((size_t)b * SEQ + q) * DM + h * HD + dd] = (__bf16)(oacc[jd][r] * inv);
    }
  }
}

// ---------------------------------------------------------------------------
// K3: output projection y = O Wo^T (einsum bsd,ed->bse), f32 out
__global__ __launch_bounds__(128) void out_proj(
    const __bf16* __restrict__ Obf, const __bf16* __restrict__ Wob,
    float* __restrict__ out)
{
  __shared__ __align__(16) __bf16 Wlds[2][64][32];
  const int tid    = threadIdx.x;
  const int lane   = tid & 31;
  const int wave   = tid >> 5;
  const int m_base = blockIdx.x * 64 + wave * 16;
  const int nb     = blockIdx.y * 64;

  const int srow  = tid >> 1;
  const int shalf = tid & 1;
  const __bf16* wsrc = Wob + (size_t)(nb + srow) * DM + shalf * 16;
  const int arow = m_base + (lane & 15);

  v8f acc[4] = {};
  wmma_gemm64(Obf + (size_t)arow * DM, wsrc, Wlds, acc, lane, srow, shalf);

  const int half = (lane >> 4) & 1;
#pragma unroll
  for (int r = 0; r < 8; ++r) {
    int m = m_base + r + 8 * half;
#pragma unroll
    for (int j = 0; j < 4; ++j)
      out[(size_t)m * DM + nb + (lane & 15) + 16 * j] = acc[j][r];
  }
}

// ---------------------------------------------------------------------------
extern "C" void kernel_launch(void* const* d_in, const int* in_sizes, int n_in,
                              void* d_out, int out_size, void* d_ws, size_t ws_size,
                              hipStream_t stream) {
  const float* x    = (const float*)d_in[0];
  const float* rcos = (const float*)d_in[1];
  const float* rsin = (const float*)d_in[2];
  const float* Wq   = (const float*)d_in[3];
  const float* Wk   = (const float*)d_in[4];
  const float* Wv   = (const float*)d_in[5];
  const float* Wo   = (const float*)d_in[6];
  float* out        = (float*)d_out;

  char* ws = (char*)d_ws;
  const size_t MB = 1024 * 1024;
  __bf16* xbf = (__bf16*)(ws);             //  8 MB  x as bf16
  __bf16* Wqb = (__bf16*)(ws +  8 * MB);   //  2 MB
  __bf16* Wkb = (__bf16*)(ws + 10 * MB);   //  2 MB
  __bf16* Wvb = (__bf16*)(ws + 12 * MB);   //  2 MB
  __bf16* Wob = (__bf16*)(ws + 14 * MB);   //  2 MB
  __bf16* Qr  = (__bf16*)(ws + 16 * MB);   //  8 MB  [B,H,S,D] rope'd
  __bf16* Kr  = (__bf16*)(ws + 24 * MB);   //  8 MB  [B,H,S,D] rope'd
  __bf16* Vt  = (__bf16*)(ws + 32 * MB);   //  8 MB  [B,H,D,S]
  __bf16* Obf = (__bf16*)(ws + 40 * MB);   //  8 MB  [B,S,DM]

  const int nx = BSZ * SEQ * DM;           // 4M
  const int nw = DM * DM;                  // 1M
  cvt_f32_bf16<<<(nx + 255) / 256, 256, 0, stream>>>(x,  xbf, nx);
  cvt_f32_bf16<<<(nw + 255) / 256, 256, 0, stream>>>(Wq, Wqb, nw);
  cvt_f32_bf16<<<(nw + 255) / 256, 256, 0, stream>>>(Wk, Wkb, nw);
  cvt_f32_bf16<<<(nw + 255) / 256, 256, 0, stream>>>(Wv, Wvb, nw);
  cvt_f32_bf16<<<(nw + 255) / 256, 256, 0, stream>>>(Wo, Wob, nw);

  qkv_proj<<<dim3(64, NH, 3), 128, 0, stream>>>(xbf, Wqb, Wkb, Wvb,
                                                rcos, rsin, Qr, Kr, Vt);
  attn<<<dim3(SEQ / 64, BSZ * NH), 128, 0, stream>>>(Qr, Kr, Vt, Obf);
  out_proj<<<dim3(64, DM / 64), 128, 0, stream>>>(Obf, Wob, out);
}
